// GraphTransformerActorCritic_51376398795499
// MI455X (gfx1250) — matrix operand order repaired and checked by reading, not compiled, not run on hardware
//
#include <hip/hip_runtime.h>
#include <hip/hip_bf16.h>
#include <math.h>

typedef __bf16 v16bf __attribute__((ext_vector_type(16)));
typedef float  v8f   __attribute__((ext_vector_type(8)));

namespace {
constexpr int Gn  = 16;     // graphs
constexpr int NPG = 512;    // nodes per graph
constexpr int Nn  = 8192;   // total nodes
constexpr int En  = 131072; // edges
constexpr int NF  = 16;     // input node features
constexpr int Dm  = 192;    // d_model
constexpr int Hh  = 8;      // heads
constexpr int DK  = 24;     // head dim
constexpr int DFF = 512;    // ffn dim
constexpr int EF  = 4;      // edge features
constexpr int Ln  = 6;      // layers
}

// ---------------------------------------------------------------------------
// Batched GEMM with fully compile-time geometry:
//   C[b] = act(alpha * A[b] x B[b] + bias) (+= if ADD)
// A: MxK row-major (LDA). B: KxN row-major (LDB), or if BT element (k,n) read
// as B[n*LDB + k]. C: MxN (LDC). Batch b = (g,h) = (b/HD, b%HD), independent
// runtime g/h pointer strides per operand. One wave per 16x(16*NT) strip.
// bf16 WMMA (v_wmma_f32_16x16x32_bf16), f32 accumulate.
// K % 32 == 0  -> no bounds logic, immediate-offset unconditional loads.
// K < 32 (even)-> single k-step, branch-free lane-half masks.
// ---------------------------------------------------------------------------
template<int NT, int ACT, bool ADD, bool BT, int N, int K, int LDA, int LDB, int LDC, int HD>
__global__ __launch_bounds__(128)
void gemm_wmma_k(const float* __restrict__ A, const float* __restrict__ B,
                 const float* __restrict__ bias, float* __restrict__ C,
                 int M, float alpha,
                 long long sAg, long long sAh, long long sBg, long long sBh,
                 long long sCg, long long sCh)
{
    constexpr int  NSTRIPS = (N + 16 * NT - 1) / (16 * NT);
    constexpr bool EXACT   = (K % 32 == 0);
    constexpr bool FULLN   = (N % (16 * NT) == 0);

    const int lane   = threadIdx.x & 31;
    const int mtiles = M >> 4;
    const int wid    = blockIdx.x * (blockDim.x >> 5) + (threadIdx.x >> 5);
    if (wid >= mtiles * NSTRIPS) return;       // wave-uniform: EXEC stays full
    const int mtile  = wid / NSTRIPS;
    const int nstrip = wid % NSTRIPS;

    const int b  = blockIdx.z;
    const int gb = b / HD, hb = b - gb * HD;
    A += gb * sAg + hb * sAh;
    B += gb * sBg + hb * sBh;
    C += gb * sCg + hb * sCh;

    const int l15   = lane & 15;
    const int khalf = lane >> 4;
    const int row   = mtile * 16 + l15;

    const float* Arow  = A + (long long)row * LDA;          // for small-K path
    const float* Abase = Arow + (khalf << 4);               // for exact path

    int ncol[NT];
    const float* Bb[NT];   // per-tile base (meaning depends on BT/EXACT)
#pragma unroll
    for (int t = 0; t < NT; ++t) {
        ncol[t] = (nstrip * NT + t) * 16 + l15;
        const int ncl = FULLN ? ncol[t] : (ncol[t] < N ? ncol[t] : N - 1);
        if (BT)
            Bb[t] = B + (long long)ncl * LDB + (EXACT ? (khalf << 4) : 0);
        else
            Bb[t] = B + ncl + (EXACT ? (long long)(khalf << 4) * LDB : 0);
    }

    v8f acc[NT];
#pragma unroll
    for (int t = 0; t < NT; ++t)
#pragma unroll
        for (int r = 0; r < 8; ++r) acc[t][r] = 0.0f;

    for (int k0 = 0; k0 < K; k0 += 32) {
        // ---- A fragment (16x32 bf16, contiguous k = 16*khalf + element) ----
        v16bf af;
        {
            float2 w[8];
            if (EXACT) {
#pragma unroll
                for (int p = 0; p < 8; ++p)
                    w[p] = *(const float2*)(Abase + k0 + (p << 1));
#pragma unroll
                for (int p = 0; p < 8; ++p) {
                    af[2 * p]     = (__bf16)w[p].x;
                    af[2 * p + 1] = (__bf16)w[p].y;
                }
            } else {
                bool okv[8];
#pragma unroll
                for (int p = 0; p < 8; ++p) {
                    const int k = (khalf << 4) + (p << 1);
                    okv[p] = (k < K);                 // K even: pair all-or-none
                    w[p] = *(const float2*)(Arow + (okv[p] ? k : 0));
                }
#pragma unroll
                for (int p = 0; p < 8; ++p) {
                    af[2 * p]     = (__bf16)(okv[p] ? w[p].x : 0.0f);
                    af[2 * p + 1] = (__bf16)(okv[p] ? w[p].y : 0.0f);
                }
            }
        }
        // ---- B fragments + WMMA ----
#pragma unroll
        for (int t = 0; t < NT; ++t) {
            v16bf bfv;
            if (BT) {
                float2 w[8];
                if (EXACT) {
#pragma unroll
                    for (int p = 0; p < 8; ++p)
                        w[p] = *(const float2*)(Bb[t] + k0 + (p << 1));
#pragma unroll
                    for (int p = 0; p < 8; ++p) {
                        bfv[2 * p]     = (__bf16)w[p].x;
                        bfv[2 * p + 1] = (__bf16)w[p].y;
                    }
                } else {
                    bool okv[8];
#pragma unroll
                    for (int p = 0; p < 8; ++p) {
                        const int k = (khalf << 4) + (p << 1);
                        okv[p] = (k < K);
                        w[p] = *(const float2*)(Bb[t] + (okv[p] ? k : 0));
                    }
#pragma unroll
                    for (int p = 0; p < 8; ++p) {
                        bfv[2 * p]     = (__bf16)(okv[p] ? w[p].x : 0.0f);
                        bfv[2 * p + 1] = (__bf16)(okv[p] ? w[p].y : 0.0f);
                    }
                }
            } else {
                float w[16];
                if (EXACT) {
                    const float* bp = Bb[t] + (long long)k0 * LDB;
#pragma unroll
                    for (int e = 0; e < 16; ++e) w[e] = bp[e * LDB];
#pragma unroll
                    for (int e = 0; e < 16; ++e) bfv[e] = (__bf16)w[e];
                } else {
                    bool okv[16];
#pragma unroll
                    for (int e = 0; e < 16; ++e) {
                        const int k = (khalf << 4) + e;
                        okv[e] = (k < K);
                        w[e] = Bb[t][(okv[e] ? k : 0) * LDB];
                    }
#pragma unroll
                    for (int e = 0; e < 16; ++e)
                        bfv[e] = (__bf16)(okv[e] ? w[e] : 0.0f);
                }
            }
            acc[t] = __builtin_amdgcn_wmma_f32_16x16x32_bf16(
                false, af, false, bfv, (short)0, acc[t], false, false);
        }
    }

    // ---- epilogue: scale, bias, activation, (add-)store ----
#pragma unroll
    for (int t = 0; t < NT; ++t) {
        if (!FULLN && ncol[t] >= N) continue;
        const float bv = bias ? bias[ncol[t]] : 0.0f;
        float* cp = C + (long long)(mtile * 16 + (khalf << 3)) * LDC + ncol[t];
#pragma unroll
        for (int r = 0; r < 8; ++r) {
            float v = acc[t][r] * alpha + bv;
            if (ACT == 1) v = v > 0.0f ? v : 0.0f;
            if (ACT == 2) v = 0.5f * v * (1.0f + erff(v * 0.70710678118654752f));
            if (ADD) cp[r * LDC] += v; else cp[r * LDC] = v;
        }
    }
}

// ---------------------------------------------------------------------------
// LayerNorm over rows of width Dm. One 256-thread block per row.
// ---------------------------------------------------------------------------
__global__ void ln_kernel(const float* __restrict__ in, const float* __restrict__ g,
                          const float* __restrict__ bta, float* __restrict__ out)
{
    __shared__ float red[256];
    const int row = blockIdx.x;
    const int tid = threadIdx.x;
    float v = (tid < Dm) ? in[(long long)row * Dm + tid] : 0.0f;
    red[tid] = v;
    __syncthreads();
    for (int s = 128; s > 0; s >>= 1) { if (tid < s) red[tid] += red[tid + s]; __syncthreads(); }
    const float mu = red[0] * (1.0f / Dm);
    __syncthreads();
    float d = (tid < Dm) ? (v - mu) : 0.0f;
    red[tid] = d * d;
    __syncthreads();
    for (int s = 128; s > 0; s >>= 1) { if (tid < s) red[tid] += red[tid + s]; __syncthreads(); }
    const float var = red[0] * (1.0f / Dm);
    if (tid < Dm)
        out[(long long)row * Dm + tid] = (v - mu) * rsqrtf(var + 1e-5f) * g[tid] + bta[tid];
}

// ---------------------------------------------------------------------------
// Scatter edge bias into scores: S[(g*H+h)*512*512 + i*512 + j] += be[e*H+h]
// ---------------------------------------------------------------------------
__global__ void scatter_bias_kernel(const int* __restrict__ ei,
                                    const float* __restrict__ be,
                                    float* __restrict__ S)
{
    const int e = blockIdx.x * blockDim.x + threadIdx.x;
    if (e >= En) return;
    const int src = ei[e], dst = ei[En + e];
    const int g = src >> 9, i = src & 511, j = dst & 511;
#pragma unroll
    for (int hh = 0; hh < Hh; ++hh)
        atomicAdd(S + (((long long)(g * Hh + hh) * 512 + i) * 512 + j),
                  be[(long long)e * Hh + hh]);
}

// ---------------------------------------------------------------------------
// Softmax over rows of 512. One 256-thread block per row (2 elems/thread).
// ---------------------------------------------------------------------------
__global__ void softmax_kernel(float* __restrict__ S)
{
    __shared__ float red[256];
    const long long row = blockIdx.x;
    const int tid = threadIdx.x;
    float* p = S + row * 512;
    const float a = p[tid], b = p[tid + 256];
    red[tid] = fmaxf(a, b);
    __syncthreads();
    for (int s = 128; s > 0; s >>= 1) { if (tid < s) red[tid] = fmaxf(red[tid], red[tid + s]); __syncthreads(); }
    const float m = red[0];
    __syncthreads();
    const float e0 = expf(a - m), e1 = expf(b - m);
    red[tid] = e0 + e1;
    __syncthreads();
    for (int s = 128; s > 0; s >>= 1) { if (tid < s) red[tid] += red[tid + s]; __syncthreads(); }
    const float inv = 1.0f / red[0];
    p[tid] = e0 * inv;
    p[tid + 256] = e1 * inv;
}

// ---------------------------------------------------------------------------
// Host-side launch helper
// ---------------------------------------------------------------------------
template<int NT, int ACT, bool ADD, bool BT, int N, int K, int LDA, int LDB, int LDC, int HD = 1>
static void launch_gemm(hipStream_t s, const float* A, const float* B, const float* bias,
                        float* C, int M, float alpha = 1.0f, int batch = 1,
                        long long sAg = 0, long long sAh = 0, long long sBg = 0,
                        long long sBh = 0, long long sCg = 0, long long sCh = 0)
{
    constexpr int NSTRIPS = (N + 16 * NT - 1) / (16 * NT);
    const int waves = (M / 16) * NSTRIPS;
    dim3 grid((waves + 3) / 4, 1, batch);
    gemm_wmma_k<NT, ACT, ADD, BT, N, K, LDA, LDB, LDC, HD><<<grid, dim3(128), 0, s>>>(
        A, B, bias, C, M, alpha, sAg, sAh, sBg, sBh, sCg, sCh);
}

extern "C" void kernel_launch(void* const* d_in, const int* in_sizes, int n_in,
                              void* d_out, int out_size, void* d_ws, size_t ws_size,
                              hipStream_t stream)
{
    const float* x     = (const float*)d_in[0];
    const int*   ei    = (const int*)  d_in[1];
    const float* ea    = (const float*)d_in[2];
    const float* projW = (const float*)d_in[3];
    const float* projb = (const float*)d_in[4];
    const float* ln0g  = (const float*)d_in[5];
    const float* ln0b  = (const float*)d_in[6];
    const float* Wq = (const float*)d_in[7],  *bq = (const float*)d_in[8];
    const float* Wk = (const float*)d_in[9],  *bk = (const float*)d_in[10];
    const float* Wv = (const float*)d_in[11], *bv = (const float*)d_in[12];
    const float* Wo = (const float*)d_in[13], *bo = (const float*)d_in[14];
    const float* eW1 = (const float*)d_in[15], *eb1 = (const float*)d_in[16];
    const float* eW2 = (const float*)d_in[17], *eb2 = (const float*)d_in[18];
    const float* ln1g = (const float*)d_in[19], *ln1b = (const float*)d_in[20];
    const float* ln2g = (const float*)d_in[21], *ln2b = (const float*)d_in[22];
    const float* fW1 = (const float*)d_in[23], *fb1 = (const float*)d_in[24];
    const float* fW2 = (const float*)d_in[25], *fb2 = (const float*)d_in[26];

    float* h  = (float*)d_out;          // running residual stream [N, D]
    float* ws = (float*)d_ws;
    float* hn = ws;                                   // [N, D]
    float* Qb = hn + (long long)Nn * Dm;              // [N, D]
    float* Kb = Qb + (long long)Nn * Dm;              // [N, D]
    float* Vb = Kb + (long long)Nn * Dm;              // [N, D]
    float* be = Vb + (long long)Nn * Dm;              // [E, H]
    float* S  = be + (long long)En * Hh;              // [G*H, 512, 512] (also edge-hidden / ffn-hidden)

    const float inv_scale = 0.20412414523193154f;     // 1/sqrt(24)

    // node projection -> Qb (tmp), then LN0 -> h
    launch_gemm<4, 0, false, false, Dm, NF, NF, Dm, Dm>(stream, x, projW, projb, Qb, Nn);
    ln_kernel<<<Nn, 256, 0, stream>>>(Qb, ln0g, ln0b, h);

    for (int l = 0; l < Ln; ++l) {
        const long long dd = (long long)l * Dm * Dm;
        // pre-norm attention input
        ln_kernel<<<Nn, 256, 0, stream>>>(h, ln1g + l * Dm, ln1b + l * Dm, hn);
        // Q, K, V projections (K=192)
        launch_gemm<4, 0, false, false, Dm, Dm, Dm, Dm, Dm>(stream, hn, Wq + dd, bq + l * Dm, Qb, Nn);
        launch_gemm<4, 0, false, false, Dm, Dm, Dm, Dm, Dm>(stream, hn, Wk + dd, bk + l * Dm, Kb, Nn);
        launch_gemm<4, 0, false, false, Dm, Dm, Dm, Dm, Dm>(stream, hn, Wv + dd, bv + l * Dm, Vb, Nn);
        // edge bias MLP: t = relu(ea@eW1+eb1) in S region (K=4), be = t@eW2+eb2 (K=192, N=8)
        launch_gemm<4, 1, false, false, Dm, EF, EF, Dm, Dm>(stream, ea, eW1 + (long long)l * EF * Dm,
                                                            eb1 + l * Dm, S, En);
        launch_gemm<1, 0, false, false, Hh, Dm, Dm, Hh, Hh>(stream, S, eW2 + (long long)l * Dm * Hh,
                                                            eb2 + l * Hh, be, En);
        // scores = Q K^T / sqrt(dk), batched over (g,h)  (K=24, B transposed)
        launch_gemm<4, 0, false, true, NPG, DK, Dm, Dm, NPG, Hh>(
            stream, Qb, Kb, nullptr, S, NPG, inv_scale, Gn * Hh,
            (long long)NPG * Dm, DK,
            (long long)NPG * Dm, DK,
            (long long)Hh * NPG * NPG, (long long)NPG * NPG);
        // scatter edge bias, softmax over keys
        scatter_bias_kernel<<<En / 256, 256, 0, stream>>>(ei, be, S);
        softmax_kernel<<<Gn * Hh * NPG, 256, 0, stream>>>(S);
        // O = attn @ V -> hn (reused as O buffer), batched over (g,h)  (K=512, N=24)
        launch_gemm<2, 0, false, false, DK, NPG, NPG, Dm, Dm, Hh>(
            stream, S, Vb, nullptr, hn, NPG, 1.0f, Gn * Hh,
            (long long)Hh * NPG * NPG, (long long)NPG * NPG,
            (long long)NPG * Dm, DK,
            (long long)NPG * Dm, DK);
        // h += O @ Wo + bo  (K=192)
        launch_gemm<4, 0, true, false, Dm, Dm, Dm, Dm, Dm>(stream, hn, Wo + dd, bo + l * Dm, h, Nn);
        // pre-norm FFN: h2 -> Qb (reused), u = gelu(h2@fW1+fb1) (K=192), h += u@fW2+fb2 (K=512)
        ln_kernel<<<Nn, 256, 0, stream>>>(h, ln2g + l * Dm, ln2b + l * Dm, Qb);
        launch_gemm<4, 2, false, false, DFF, Dm, Dm, DFF, DFF>(stream, Qb, fW1 + (long long)l * Dm * DFF,
                                                               fb1 + l * DFF, S, Nn);
        launch_gemm<4, 0, true, false, Dm, DFF, DFF, Dm, Dm>(stream, S, fW2 + (long long)l * DFF * Dm,
                                                             fb2 + l * Dm, h, Nn);
    }
}
